// Autoencoder_14972255994442
// MI455X (gfx1250) — compile-verified, matrix operand order checked by reading, not run on hardware
//
#include <hip/hip_runtime.h>
#include <hip/hip_bf16.h>
#include <stdint.h>
#include <math.h>

// ---------------------------------------------------------------------------
// Types for CDNA5 WMMA (wave32): bf16 A/B fragments, f32 accumulator.
// ---------------------------------------------------------------------------
typedef __attribute__((ext_vector_type(16))) __bf16 bf16x16;
typedef __attribute__((ext_vector_type(8)))  __bf16 bf16x8;
typedef __attribute__((ext_vector_type(8)))  float  f32x8;

#define S_LEN  48
#define TSTEPS 49            // S+1
#define BB     16
#define SPB    4
#define NN     64            // B*SPB
#define E2D    1024
#define HD     1024
#define HhD    512
#define VV     16003
#define VP     16016         // V padded to multiple of 16
#define MEMD   500
#define MEMP   512
#define MROWS_PAD 832        // 784 mem-MLP rows padded to multiple of 64

__device__ __forceinline__ float sigf(float x) { return 1.0f / (1.0f + __expf(-x)); }

__device__ __forceinline__ unsigned hash3(unsigned a, unsigned b, unsigned c) {
    unsigned x = a * 0x9E3779B1u ^ b * 0x85EBCA77u ^ c * 0xC2B2AE3Du;
    x ^= x >> 16; x *= 0x7FEB352Du;
    x ^= x >> 15; x *= 0x846CA68Bu;
    x ^= x >> 16;
    return x;
}
__device__ __forceinline__ float u01(unsigned h) {
    return ((h >> 8) + 0.5f) * (1.0f / 16777216.0f);
}

// ---------------------------------------------------------------------------
// Core GEMM: C[M x Ncp] = A[M x K](bf16) * W[Ncp x K]^T(bf16) (+bias)(+relu)
// One wave computes a 64x16 strip (4 M-tiles x 1 N-tile): each B fragment is
// reused by 4 v_wmma_f32_16x16x32_bf16, amortizing the dominant weight stream.
// A-frag: lane L<16 -> row L, K chunks {k0+0..7, k0+16..23}; lanes 16..31 -> +8.
// B-frag identical with W rows (NT gemm). D: acc[r] -> C[m0+r+8*hs][n].
// A must be allocated with rows padded up to a multiple of 64 (reads only).
// ---------------------------------------------------------------------------
__global__ __launch_bounds__(128)
void gemm_bf16_wmma(const __bf16* __restrict__ A, const __bf16* __restrict__ W,
                    const float* __restrict__ bias, float* __restrict__ C,
                    __bf16* __restrict__ Cbf,
                    int M, int Ncp, int Nc, int K, int beta, int relu)
{
    const int lane = threadIdx.x & 31;
    const int wave = threadIdx.x >> 5;
    const int tn   = blockIdx.x * 4 + wave;          // column tile (wave-uniform)
    const int tm0  = blockIdx.y * 4;                  // first of 4 row tiles
    if (tn * 16 >= Ncp) return;                       // uniform per wave: EXEC stays full

    const int r16 = lane & 15;
    const int hs  = lane >> 4;
    const int n   = tn * 16 + r16;

    const __bf16* __restrict__ wrow = W + (size_t)n * K;
    const __bf16* arow[4];
#pragma unroll
    for (int mt = 0; mt < 4; ++mt)
        arow[mt] = A + (size_t)((tm0 + mt) * 16 + r16) * K;

    f32x8 acc[4];
    if (beta) {
#pragma unroll
        for (int mt = 0; mt < 4; ++mt)
#pragma unroll
            for (int r = 0; r < 8; ++r) {
                int row = (tm0 + mt) * 16 + r + 8 * hs;
                acc[mt][r] = (row < M) ? C[(size_t)row * Ncp + n] : 0.0f;
            }
    } else {
        float bv = (bias != nullptr && n < Nc) ? bias[n] : 0.0f;
#pragma unroll
        for (int mt = 0; mt < 4; ++mt)
#pragma unroll
            for (int r = 0; r < 8; ++r) acc[mt][r] = bv;
    }

    for (int k0 = 0; k0 < K; k0 += 32) {
        bf16x8 blo = *(const bf16x8*)(wrow + k0 + hs * 8);
        bf16x8 bhi = *(const bf16x8*)(wrow + k0 + 16 + hs * 8);
        bf16x16 b = __builtin_shufflevector(blo, bhi, 0,1,2,3,4,5,6,7,8,9,10,11,12,13,14,15);
#pragma unroll
        for (int mt = 0; mt < 4; ++mt) {
            bf16x8 alo = *(const bf16x8*)(arow[mt] + k0 + hs * 8);
            bf16x8 ahi = *(const bf16x8*)(arow[mt] + k0 + 16 + hs * 8);
            bf16x16 a = __builtin_shufflevector(alo, ahi,
                                                0,1,2,3,4,5,6,7,8,9,10,11,12,13,14,15);
            acc[mt] = __builtin_amdgcn_wmma_f32_16x16x32_bf16(false, a, false, b,
                                                              (short)0, acc[mt],
                                                              false, false);
        }
    }

#pragma unroll
    for (int mt = 0; mt < 4; ++mt) {
#pragma unroll
        for (int r = 0; r < 8; ++r) {
            float v = relu ? fmaxf(acc[mt][r], 0.0f) : acc[mt][r];
            int row = (tm0 + mt) * 16 + r + 8 * hs;
            if (row < M && n < Nc) {
                C[(size_t)row * Ncp + n] = v;
                if (Cbf) Cbf[(size_t)row * Ncp + n] = (__bf16)v;
            }
        }
    }
}

// ---------------------------------------------------------------------------
// Elementwise / utility kernels
// ---------------------------------------------------------------------------
__global__ void fill_f32(float* p, float v, long n) {
    long i = (long)blockIdx.x * blockDim.x + threadIdx.x;
    if (i < n) p[i] = v;
}
__global__ void fill_bf16(__bf16* p, float v, long n) {
    long i = (long)blockIdx.x * blockDim.x + threadIdx.x;
    if (i < n) p[i] = (__bf16)v;
}
__global__ void fill_i32(int* p, int v, long n) {
    long i = (long)blockIdx.x * blockDim.x + threadIdx.x;
    if (i < n) p[i] = v;
}
__global__ void cvt_bf16_k(const float* __restrict__ s, __bf16* __restrict__ d, long n) {
    long i = (long)blockIdx.x * blockDim.x + threadIdx.x;
    if (i < n) d[i] = (__bf16)s[i];
}
// dst is padRows x cols; rows >= real rows get zero
__global__ void cvt_pad_rows(const float* __restrict__ s, __bf16* __restrict__ d,
                             int rows, int cols, int padRows) {
    long i = (long)blockIdx.x * blockDim.x + threadIdx.x;
    long tot = (long)padRows * cols;
    if (i >= tot) return;
    int r = (int)(i / cols);
    d[i] = (r < rows) ? (__bf16)s[i] : (__bf16)0.0f;
}
__global__ void add_vec(const float* a, const float* b, float* o, int n) {
    int i = blockIdx.x * blockDim.x + threadIdx.x;
    if (i < n) o[i] = a[i] + b[i];
}
__global__ void build_num(const int* __restrict__ numeric, int* __restrict__ num) {
    int i = blockIdx.x * blockDim.x + threadIdx.x;
    if (i >= TSTEPS * BB) return;
    int s = i / BB, b = i - s * BB;
    num[i] = (s == 0) ? 0 : numeric[(s - 1) * BB + b];
}
__global__ void gather_embed(const float* __restrict__ embed, const int* __restrict__ idx,
                             __bf16* __restrict__ dst, int rows, int cols) {
    long i = (long)blockIdx.x * blockDim.x + threadIdx.x;
    if (i >= (long)rows * cols) return;
    int r = (int)(i / cols), c = (int)(i - (long)r * cols);
    dst[i] = (__bf16)embed[(size_t)idx[r] * cols + c];
}
// sigmoid head of the mem-MLP + bernoulli noising of the token stream
__global__ void mem_head_noise(const float* __restrict__ hmem, const float* __restrict__ w2,
                               const float* __restrict__ b2, const int* __restrict__ num,
                               int* __restrict__ num_noised) {
    int i = blockIdx.x * blockDim.x + threadIdx.x;
    if (i >= TSTEPS * BB) return;
    float acc = b2[0];
    const float* hr = hmem + (size_t)i * MEMP;
    for (int m = 0; m < MEMD; ++m) acc += hr[m] * w2[m];
    float mh = sigf(acc);
    int w = num[i];
    int f = (u01(hash3(0x6D656D01u, (unsigned)i, 1u)) < mh) ? 1 : 0;
    if (w == 3) f = 1;                               // PERIOD always kept
    num_noised[i] = f ? w : 0;
}
__global__ void rep_kernel(const int* __restrict__ num, const int* __restrict__ num_noised,
                           int* __restrict__ numr, int* __restrict__ noisr) {
    int i = blockIdx.x * blockDim.x + threadIdx.x;
    if (i >= TSTEPS * NN) return;
    int s = i / NN, n = i - s * NN;
    int src = s * BB + n / SPB;
    numr[i]  = num[src];
    noisr[i] = num_noised[src];
}

// LSTM cell pointwise: z[N x 4*hid] -> (h,c); h written as bf16 state and
// optionally into f32/bf16 sequence buffers at (out_row0+n)*stride + off + j.
__global__ void lstm_cell(const float* __restrict__ z, float* __restrict__ c,
                          __bf16* __restrict__ h_bf, int hid,
                          float* __restrict__ outF, __bf16* __restrict__ outBF,
                          int out_row0, int out_stride, int out_off) {
    int i = blockIdx.x * blockDim.x + threadIdx.x;
    if (i >= NN * hid) return;
    int nIdx = i / hid, j = i - nIdx * hid;
    const float* zr = z + (size_t)nIdx * 4 * hid;
    float zi = zr[j], zf = zr[hid + j], zg = zr[2 * hid + j], zo = zr[3 * hid + j];
    float cn = sigf(zf) * c[i] + sigf(zi) * tanhf(zg);
    float hn = sigf(zo) * tanhf(cn);
    c[i] = cn;
    h_bf[i] = (__bf16)hn;
    size_t oi = (size_t)(out_row0 + nIdx) * out_stride + out_off + j;
    if (outF)  outF[oi]  = hn;
    if (outBF) outBF[oi] = (__bf16)hn;
}

// scores[s,n] = dot(proj_enc[s*NN+n,:], hdec[n,:])
__global__ void scores_kernel(const float* __restrict__ proj_enc,
                              const float* __restrict__ hdec, float* __restrict__ scores) {
    int i = blockIdx.x * blockDim.x + threadIdx.x;
    if (i >= TSTEPS * NN) return;
    int n = i % NN;
    const float* p = proj_enc + (size_t)i * HD;
    const float* h = hdec + (size_t)n * HD;
    float acc = 0.0f;
    for (int k = 0; k < HD; ++k) acc += p[k] * h[k];
    scores[i] = acc;
}
__global__ void softmax_s(const float* __restrict__ scores, float* __restrict__ attn) {
    int n = blockIdx.x * blockDim.x + threadIdx.x;
    if (n >= NN) return;
    float m = -1e30f;
    for (int s = 0; s < TSTEPS; ++s) m = fmaxf(m, scores[s * NN + n]);
    float sum = 0.0f;
    for (int s = 0; s < TSTEPS; ++s) sum += __expf(scores[s * NN + n] - m);
    float inv = 1.0f / sum;
    for (int s = 0; s < TSTEPS; ++s) attn[s * NN + n] = __expf(scores[s * NN + n] - m) * inv;
}
__global__ void attn_ctx(const float* __restrict__ attn, const float* __restrict__ out_enc,
                         float* __restrict__ ctx) {
    int i = blockIdx.x * blockDim.x + threadIdx.x;
    if (i >= NN * HD) return;
    int n = i / HD, h = i - n * HD;
    float acc = 0.0f;
    for (int s = 0; s < TSTEPS; ++s)
        acc += attn[s * NN + n] * out_enc[((size_t)s * NN + n) * HD + h];
    ctx[i] = acc;
}
__global__ void concat_full(const float* __restrict__ hdec, const float* __restrict__ ctx,
                            __bf16* __restrict__ full) {
    int i = blockIdx.x * blockDim.x + threadIdx.x;
    if (i >= NN * 2 * HD) return;
    int n = i / (2 * HD), j = i - n * 2 * HD;
    float v = (j < HD) ? hdec[(size_t)n * HD + j] : ctx[(size_t)n * HD + (j - HD)];
    full[i] = (__bf16)v;
}

// Per-row (n) categorical sample via gumbel-argmax + log_softmax gather.
__global__ __launch_bounds__(256)
void sample_kernel(const float* __restrict__ logits, const int* __restrict__ noisr,
                   const int* __restrict__ numr, int t, int* __restrict__ next_word,
                   float* __restrict__ out_lp, float* __restrict__ out_res) {
    __shared__ float sred[256];
    __shared__ int   sidx[256];
    const int n = blockIdx.x, tid = threadIdx.x;
    const float* lg = logits + (size_t)n * VP;

    float m = -1e30f;
    for (int v = tid; v < VV; v += 256) m = fmaxf(m, lg[v]);
    sred[tid] = m; __syncthreads();
    for (int off = 128; off > 0; off >>= 1) {
        if (tid < off) sred[tid] = fmaxf(sred[tid], sred[tid + off]);
        __syncthreads();
    }
    float mx = sred[0]; __syncthreads();

    float sum = 0.0f;
    for (int v = tid; v < VV; v += 256) sum += __expf(lg[v] - mx);
    sred[tid] = sum; __syncthreads();
    for (int off = 128; off > 0; off >>= 1) {
        if (tid < off) sred[tid] += sred[tid + off];
        __syncthreads();
    }
    float logZ = mx + logf(sred[0]); __syncthreads();

    float best = -1e30f; int bi = 0;
    for (int v = tid; v < VV; v += 256) {
        float u = u01(hash3(0x73616D00u + (unsigned)t, (unsigned)n, (unsigned)v));
        float g = (lg[v] - logZ) - logf(-logf(u));
        if (g > best) { best = g; bi = v; }
    }
    sred[tid] = best; sidx[tid] = bi; __syncthreads();
    for (int off = 128; off > 0; off >>= 1) {
        if (tid < off) {
            if (sred[tid + off] > sred[tid] ||
                (sred[tid + off] == sred[tid] && sidx[tid + off] < sidx[tid])) {
                sred[tid] = sred[tid + off]; sidx[tid] = sidx[tid + off];
            }
        }
        __syncthreads();
    }
    if (tid == 0) {
        int sample = sidx[0];
        int noi = noisr[t * NN + n], nm = numr[t * NN + n];
        int nw = (noi == 0) ? sample : nm;
        next_word[n] = nw;
        out_lp[t * NN + n]  = lg[nw] - logZ;
        out_res[t * NN + n] = (float)nw;
    }
}

// ---------------------------------------------------------------------------
// Host orchestration
// ---------------------------------------------------------------------------
extern "C" void kernel_launch(void* const* d_in, const int* in_sizes, int n_in,
                              void* d_out, int out_size, void* d_ws, size_t ws_size,
                              hipStream_t stream)
{
    (void)in_sizes; (void)n_in; (void)out_size; (void)ws_size;

    const int*   numeric = (const int*)  d_in[0];
    const float* embed   = (const float*)d_in[1];
    const float* mem_W1  = (const float*)d_in[2];
    const float* mem_b1  = (const float*)d_in[3];
    const float* mem_W2  = (const float*)d_in[4];
    const float* mem_b2  = (const float*)d_in[5];
    const float *eWih[4], *eWhh[4], *eBih[4], *eBhh[4];
    for (int l = 0; l < 2; ++l)
        for (int d = 0; d < 2; ++d) {
            int base = 6 + (l * 2 + d) * 4, i = l * 2 + d;
            eWih[i] = (const float*)d_in[base + 0];
            eWhh[i] = (const float*)d_in[base + 1];
            eBih[i] = (const float*)d_in[base + 2];
            eBhh[i] = (const float*)d_in[base + 3];
        }
    const float *dWih[2], *dWhh[2], *dBih[2], *dBhh[2];
    for (int l = 0; l < 2; ++l) {
        int base = 22 + l * 4;
        dWih[l] = (const float*)d_in[base + 0];
        dWhh[l] = (const float*)d_in[base + 1];
        dBih[l] = (const float*)d_in[base + 2];
        dBhh[l] = (const float*)d_in[base + 3];
    }
    const float* attn_W = (const float*)d_in[30];
    const float* mlp_W  = (const float*)d_in[31];
    const float* mlp_b  = (const float*)d_in[32];
    const float* out_W  = (const float*)d_in[33];
    const float* out_b  = (const float*)d_in[34];

    float* out_lp  = (float*)d_out;
    float* out_res = out_lp + TSTEPS * NN;

    // --- workspace bump allocator (256B aligned) ---
    char* wsb = (char*)d_ws; size_t off = 0;
    auto alloc = [&](size_t bytes) -> void* {
        void* p = wsb + off;
        off = (off + bytes + 255) & ~(size_t)255;
        return p;
    };
    // bf16 weights
    __bf16 *eWihB[4], *eWhhB[4], *dWihB[2], *dWhhB[2];
    for (int i = 0; i < 4; ++i) eWihB[i] = (__bf16*)alloc((size_t)4 * HhD * E2D * 2);
    for (int i = 0; i < 4; ++i) eWhhB[i] = (__bf16*)alloc((size_t)4 * HhD * HhD * 2);
    for (int i = 0; i < 2; ++i) dWihB[i] = (__bf16*)alloc((size_t)4 * HD * E2D * 2);
    for (int i = 0; i < 2; ++i) dWhhB[i] = (__bf16*)alloc((size_t)4 * HD * HD * 2);
    __bf16* attnB = (__bf16*)alloc((size_t)HD * HD * 2);
    __bf16* mlpB  = (__bf16*)alloc((size_t)HD * 2 * HD * 2);
    __bf16* outWB = (__bf16*)alloc((size_t)VP * HD * 2);
    __bf16* memW1B= (__bf16*)alloc((size_t)MEMP * E2D * 2);
    // combined biases
    float *eBc[4], *dBc[2];
    for (int i = 0; i < 4; ++i) eBc[i] = (float*)alloc(4 * HhD * 4);
    for (int i = 0; i < 2; ++i) dBc[i] = (float*)alloc(4 * HD * 4);
    // token / noise buffers
    int* num        = (int*)alloc(TSTEPS * BB * 4);
    int* num_noised = (int*)alloc(TSTEPS * BB * 4);
    int* numr       = (int*)alloc(TSTEPS * NN * 4);
    int* noisr      = (int*)alloc(TSTEPS * NN * 4);
    // mem MLP (A rows padded to multiple of 64 for the 4x M-blocked GEMM)
    __bf16* embAllB = (__bf16*)alloc((size_t)MROWS_PAD * E2D * 2);
    float*  hmem    = (float*)alloc((size_t)TSTEPS * BB * MEMP * 4);
    // encoder sequence buffers (ping/pong bf16, final f32)
    __bf16* seqA   = (__bf16*)alloc((size_t)TSTEPS * NN * HD * 2);
    __bf16* seqBuf = (__bf16*)alloc((size_t)TSTEPS * NN * HD * 2);
    float*  out_enc= (float*) alloc((size_t)TSTEPS * NN * HD * 4);
    float*  projE  = (float*) alloc((size_t)TSTEPS * NN * HD * 4);
    // shared z buffer (max 64 x 4096)
    float* zbuf = (float*)alloc((size_t)NN * 4 * HD * 4);
    // encoder state
    __bf16* ehB = (__bf16*)alloc((size_t)NN * HhD * 2);
    float*  ec  = (float*) alloc((size_t)NN * HhD * 4);
    // decoder state
    __bf16* h1B = (__bf16*)alloc((size_t)NN * HD * 2);
    __bf16* h2B = (__bf16*)alloc((size_t)NN * HD * 2);
    float*  c1  = (float*) alloc((size_t)NN * HD * 4);
    float*  c2  = (float*) alloc((size_t)NN * HD * 4);
    float*  h2F = (float*) alloc((size_t)NN * HD * 4);
    float*  sco = (float*) alloc(TSTEPS * NN * 4);
    float*  atn = (float*) alloc(TSTEPS * NN * 4);
    float*  ctx = (float*) alloc((size_t)NN * HD * 4);
    __bf16* fullB   = (__bf16*)alloc((size_t)NN * 2 * HD * 2);
    float*  hidden  = (float*) alloc((size_t)NN * HD * 4);
    __bf16* hiddenB = (__bf16*)alloc((size_t)NN * HD * 2);
    float*  logits  = (float*) alloc((size_t)NN * VP * 4);
    __bf16* embLast = (__bf16*)alloc((size_t)NN * E2D * 2);
    int*    nextW   = (int*)   alloc(NN * 4);

    auto cvt = [&](const float* s, __bf16* d, long nEl) {
        cvt_bf16_k<<<(unsigned)((nEl + 255) / 256), 256, 0, stream>>>(s, d, nEl);
    };
    auto gemm = [&](const __bf16* A, const __bf16* W, const float* bias, float* C,
                    __bf16* Cbf, int M, int Ncp, int Nc, int K, int beta, int relu) {
        dim3 g((Ncp / 16 + 3) / 4, (M / 16 + 3) / 4);
        gemm_bf16_wmma<<<g, 128, 0, stream>>>(A, W, bias, C, Cbf,
                                              M, Ncp, Nc, K, beta, relu);
    };

    // --- 1. weight precision conversion (once per launch) ---
    for (int i = 0; i < 4; ++i) cvt(eWih[i], eWihB[i], (long)4 * HhD * E2D);
    for (int i = 0; i < 4; ++i) cvt(eWhh[i], eWhhB[i], (long)4 * HhD * HhD);
    for (int i = 0; i < 2; ++i) cvt(dWih[i], dWihB[i], (long)4 * HD * E2D);
    for (int i = 0; i < 2; ++i) cvt(dWhh[i], dWhhB[i], (long)4 * HD * HD);
    cvt(attn_W, attnB, (long)HD * HD);
    cvt(mlp_W,  mlpB,  (long)HD * 2 * HD);
    cvt_pad_rows<<<(unsigned)(((long)VP * HD + 255) / 256), 256, 0, stream>>>(
        out_W, outWB, VV, HD, VP);
    cvt_pad_rows<<<(unsigned)(((long)MEMP * E2D + 255) / 256), 256, 0, stream>>>(
        mem_W1, memW1B, MEMD, E2D, MEMP);
    for (int i = 0; i < 4; ++i)
        add_vec<<<(4 * HhD + 255) / 256, 256, 0, stream>>>(eBih[i], eBhh[i], eBc[i], 4 * HhD);
    for (int i = 0; i < 2; ++i)
        add_vec<<<(4 * HD + 255) / 256, 256, 0, stream>>>(dBih[i], dBhh[i], dBc[i], 4 * HD);

    // --- 2. token stream + mem-MLP noising ---
    build_num<<<(TSTEPS * BB + 255) / 256, 256, 0, stream>>>(numeric, num);
    gather_embed<<<(unsigned)(((long)TSTEPS * BB * E2D + 255) / 256), 256, 0, stream>>>(
        embed, num, embAllB, TSTEPS * BB, E2D);
    gemm(embAllB, memW1B, mem_b1, hmem, nullptr, TSTEPS * BB, MEMP, MEMD, E2D, 0, 1);
    mem_head_noise<<<(TSTEPS * BB + 255) / 256, 256, 0, stream>>>(
        hmem, mem_W2, mem_b2, num, num_noised);
    rep_kernel<<<(TSTEPS * NN + 255) / 256, 256, 0, stream>>>(num, num_noised, numr, noisr);

    // --- 3. encoder: 2 bidirectional LSTM layers (WMMA gate GEMMs) ---
    gather_embed<<<(unsigned)(((long)TSTEPS * NN * E2D + 255) / 256), 256, 0, stream>>>(
        embed, noisr, seqA, TSTEPS * NN, E2D);
    for (int layer = 0; layer < 2; ++layer) {
        const __bf16* X   = (layer == 0) ? seqA : seqBuf;
        __bf16* outBF     = (layer == 0) ? seqBuf : seqA;
        float*  outF      = (layer == 0) ? nullptr : out_enc;
        for (int dir = 0; dir < 2; ++dir) {
            int pi = layer * 2 + dir;
            fill_bf16<<<(NN * HhD + 255) / 256, 256, 0, stream>>>(ehB, 0.0f, NN * HhD);
            fill_f32 <<<(NN * HhD + 255) / 256, 256, 0, stream>>>(ec,  0.0f, NN * HhD);
            for (int step = 0; step < TSTEPS; ++step) {
                int s = dir ? (TSTEPS - 1 - step) : step;
                gemm(X + (size_t)s * NN * HD, eWihB[pi], eBc[pi], zbuf, nullptr,
                     NN, 4 * HhD, 4 * HhD, E2D, 0, 0);
                gemm(ehB, eWhhB[pi], nullptr, zbuf, nullptr,
                     NN, 4 * HhD, 4 * HhD, HhD, 1, 0);
                lstm_cell<<<(NN * HhD + 255) / 256, 256, 0, stream>>>(
                    zbuf, ec, ehB, HhD, outF, outBF, s * NN, HD, dir * HhD);
            }
        }
    }
    // attention projection: proj_enc = out_enc @ attn_W^T  (M = 49*64)
    gemm(seqA, attnB, nullptr, projE, nullptr, TSTEPS * NN, HD, HD, HD, 0, 0);

    // --- 4. decoder scan ---
    fill_bf16<<<(NN * HD + 255) / 256, 256, 0, stream>>>(h1B, 0.0f, NN * HD);
    fill_bf16<<<(NN * HD + 255) / 256, 256, 0, stream>>>(h2B, 0.0f, NN * HD);
    fill_f32 <<<(NN * HD + 255) / 256, 256, 0, stream>>>(c1, 0.0f, NN * HD);
    fill_f32 <<<(NN * HD + 255) / 256, 256, 0, stream>>>(c2, 0.0f, NN * HD);
    fill_i32 <<<1, 64, 0, stream>>>(nextW, 0, NN);   // embedded[0] == embed[0]
    gather_embed<<<(unsigned)(((long)NN * E2D + 255) / 256), 256, 0, stream>>>(
        embed, nextW, embLast, NN, E2D);

    for (int t = 0; t < TSTEPS; ++t) {
        // layer 1
        gemm(embLast, dWihB[0], dBc[0], zbuf, nullptr, NN, 4 * HD, 4 * HD, E2D, 0, 0);
        gemm(h1B, dWhhB[0], nullptr, zbuf, nullptr, NN, 4 * HD, 4 * HD, HD, 1, 0);
        lstm_cell<<<(NN * HD + 255) / 256, 256, 0, stream>>>(
            zbuf, c1, h1B, HD, nullptr, nullptr, 0, HD, 0);
        // layer 2
        gemm(h1B, dWihB[1], dBc[1], zbuf, nullptr, NN, 4 * HD, 4 * HD, HD, 0, 0);
        gemm(h2B, dWhhB[1], nullptr, zbuf, nullptr, NN, 4 * HD, 4 * HD, HD, 1, 0);
        lstm_cell<<<(NN * HD + 255) / 256, 256, 0, stream>>>(
            zbuf, c2, h2B, HD, h2F, nullptr, 0, HD, 0);
        // attention
        scores_kernel<<<(TSTEPS * NN + 255) / 256, 256, 0, stream>>>(projE, h2F, sco);
        softmax_s<<<1, 64, 0, stream>>>(sco, atn);
        attn_ctx<<<(NN * HD + 255) / 256, 256, 0, stream>>>(atn, out_enc, ctx);
        concat_full<<<(NN * 2 * HD + 255) / 256, 256, 0, stream>>>(h2F, ctx, fullB);
        // MLP head (relu + fused bf16 output) + vocab projection
        gemm(fullB, mlpB, mlp_b, hidden, hiddenB, NN, HD, HD, 2 * HD, 0, 1);
        gemm(hiddenB, outWB, out_b, logits, nullptr, NN, VP, VV, HD, 0, 0);
        // sample + outputs + next embedding
        sample_kernel<<<NN, 256, 0, stream>>>(logits, noisr, numr, t, nextW, out_lp, out_res);
        gather_embed<<<(unsigned)(((long)NN * E2D + 255) / 256), 256, 0, stream>>>(
            embed, nextW, embLast, NN, E2D);
    }
}
